// PointNetPlusEncoder_11149735100896
// MI455X (gfx1250) — matrix operand order repaired and checked by reading, not compiled, not run on hardware
//
#include <hip/hip_runtime.h>
#include <hip/hip_bf16.h>

typedef __attribute__((ext_vector_type(16))) _Float16 v16h;
typedef __attribute__((ext_vector_type(8)))  float    v8f;

static __device__ __forceinline__ float swishf(float x) { return x / (1.f + expf(-x)); }

static inline int idiv(int a, int b) { return (a + b - 1) / b; }

// Pack two fp32 -> one dword of two f16
static __device__ __forceinline__ unsigned pk2h(float x0, float x1)
{
    union { unsigned u; _Float16 h[2]; } pk;
    pk.h[0] = (_Float16)x0; pk.h[1] = (_Float16)x1;
    return pk.u;
}

// ---------------------------------------------------------------------------
// Generic WMMA GEMM:  C[M,N] = op(A)[M,K] * op(B)[K,N] (+ bias[M])
// fp32 in/out, f16 WMMA compute. wave32: one wave per 32x32 C tile
// (2x2 fragment grid -> 4 v_wmma_f32_16x16x32_f16 per K-step, each A/B
// fragment reused twice).  LDS tiles are stored pre-swizzled in the exact
// per-lane packed fragment layout, so fragment loads are 2x ds_load_b128.
// TA/TB are compile-time so the tile-load loops are straight-line.
// Batched via blockIdx.z with element strides.
// ---------------------------------------------------------------------------
template <int TA, int TB>
__global__ void k_wmma_gemm_t(const float* __restrict__ A, const float* __restrict__ Bm,
                              float* __restrict__ C, const float* __restrict__ bias,
                              int M, int N, int K,
                              int lda, int ldb, int ldc,
                              long long sA, long long sB, long long sC)
{
    const float* Ab = A  + (long long)blockIdx.z * sA;
    const float* Bb = Bm + (long long)blockIdx.z * sB;
    float*       Cb = C  + (long long)blockIdx.z * sC;
    int tm = blockIdx.y << 5, tn = blockIdx.x << 5;
    __shared__ __align__(32) unsigned Apk[2][32][8];   // [m-subtile][lane][vgpr]
    __shared__ __align__(32) unsigned Bpk[2][32][8];   // [n-subtile][lane][vgpr]
    int lane = threadIdx.x;
    v8f acc00 = {}, acc01 = {}, acc10 = {}, acc11 = {};
    for (int k0 = 0; k0 < K; k0 += 32) {
        if (k0 + 32 < K) {  // next-tile prefetch -> global_prefetch_b8
            __builtin_prefetch(TA ? (const void*)&Ab[(long long)(k0 + 32) * lda + tm]
                                  : (const void*)&Ab[(long long)tm * lda + (k0 + 32)], 0, 1);
            __builtin_prefetch(TB ? (const void*)&Bb[(long long)tn * ldb + (k0 + 32)]
                                  : (const void*)&Bb[(long long)(k0 + 32) * ldb + tn], 0, 1);
        }
        // A tile: 32 rows x 16 k-pairs, written directly in fragment layout.
        for (int p = lane; p < 512; p += 32) {
            int m = p & 31, j = p >> 5;            // k = 2j, 2j+1
            int gm = tm + m, gk = k0 + 2 * j;
            float x0 = 0.f, x1 = 0.f;
            if (gm < M) {
                if (gk < K)     x0 = TA ? Ab[(long long)gk * lda + gm]
                                        : Ab[(long long)gm * lda + gk];
                if (gk + 1 < K) x1 = TA ? Ab[(long long)(gk + 1) * lda + gm]
                                        : Ab[(long long)gm * lda + gk + 1];
            }
            int ln = (m & 15) + (((j & 7) >= 4) ? 16 : 0);   // lane-half from (k%16)/8
            int vg = ((j >> 3) << 2) + (j & 3);              // VGPR slot
            Apk[m >> 4][ln][vg] = pk2h(x0, x1);
        }
        // B tile: 16 k-pairs x 32 cols, fragment layout.
        for (int p = lane; p < 512; p += 32) {
            int n = p & 31, j = p >> 5;
            int gn = tn + n, gk = k0 + 2 * j;
            float x0 = 0.f, x1 = 0.f;
            if (gn < N) {
                if (gk < K)     x0 = TB ? Bb[(long long)gn * ldb + gk]
                                        : Bb[(long long)gk * ldb + gn];
                if (gk + 1 < K) x1 = TB ? Bb[(long long)gn * ldb + gk + 1]
                                        : Bb[(long long)(gk + 1) * ldb + gn];
            }
            int ln = (n & 15) + (((j & 7) >= 4) ? 16 : 0);   // lane-half from (k%16)/8
            int vg = ((j >> 3) << 2) + (j & 3);
            Bpk[n >> 4][ln][vg] = pk2h(x0, x1);
        }
        __syncthreads();
        v16h a0 = *(const v16h*)(&Apk[0][lane][0]);
        v16h a1 = *(const v16h*)(&Apk[1][lane][0]);
        v16h b0 = *(const v16h*)(&Bpk[0][lane][0]);
        v16h b1 = *(const v16h*)(&Bpk[1][lane][0]);
        acc00 = __builtin_amdgcn_wmma_f32_16x16x32_f16(false, a0, false, b0, (short)0, acc00, false, false);
        acc01 = __builtin_amdgcn_wmma_f32_16x16x32_f16(false, a0, false, b1, (short)0, acc01, false, false);
        acc10 = __builtin_amdgcn_wmma_f32_16x16x32_f16(false, a1, false, b0, (short)0, acc10, false, false);
        acc11 = __builtin_amdgcn_wmma_f32_16x16x32_f16(false, a1, false, b1, (short)0, acc11, false, false);
        __syncthreads();
    }
    int ncol = lane & 15, hf = lane >> 4;
    v8f accs[2][2] = { { acc00, acc01 }, { acc10, acc11 } };
#pragma unroll
    for (int i = 0; i < 2; ++i)
#pragma unroll
        for (int j2 = 0; j2 < 2; ++j2)
#pragma unroll
            for (int q = 0; q < 8; ++q) {
                int m = tm + i * 16 + q + (hf << 3);
                int n = tn + j2 * 16 + ncol;
                if (m < M && n < N) {
                    float o = accs[i][j2][q];
                    if (bias) o += bias[m];
                    Cb[(long long)m * ldc + n] = o;
                }
            }
}

// ---------------------------------------------------------------------------
// 3x3x3 SAME conv on R^3 grid as im2col-on-the-fly WMMA GEMM.
// W: [Cout, Cin*27], X: (B, Cin, R^3), Y: (B, Cout, R^3)
// Same 32x32-per-wave / packed-LDS structure as the GEMM.
// ---------------------------------------------------------------------------
__global__ void k_conv3d_wmma(const float* __restrict__ W, const float* __restrict__ bias,
                              const float* __restrict__ X, float* __restrict__ Y,
                              int Cout, int Cin, int R)
{
    int R2 = R * R, R3 = R2 * R;
    const float* Xb = X + (long long)blockIdx.z * Cin * R3;
    float*       Yb = Y + (long long)blockIdx.z * Cout * R3;
    int tm = blockIdx.y << 5;   // cout tile
    int tn = blockIdx.x << 5;   // voxel tile
    int Kfull = Cin * 27;
    __shared__ __align__(32) unsigned Apk[2][32][8];
    __shared__ __align__(32) unsigned Bpk[2][32][8];
    int lane = threadIdx.x;
    v8f acc00 = {}, acc01 = {}, acc10 = {}, acc11 = {};
    for (int k0 = 0; k0 < Kfull; k0 += 32) {
        for (int p = lane; p < 512; p += 32) {
            int m = p & 31, j = p >> 5;
            int gm = tm + m, gk = k0 + 2 * j;
            float x0 = 0.f, x1 = 0.f;
            if (gm < Cout) {
                if (gk < Kfull)     x0 = W[(long long)gm * Kfull + gk];
                if (gk + 1 < Kfull) x1 = W[(long long)gm * Kfull + gk + 1];
            }
            int ln = (m & 15) + (((j & 7) >= 4) ? 16 : 0);
            int vg = ((j >> 3) << 2) + (j & 3);
            Apk[m >> 4][ln][vg] = pk2h(x0, x1);
        }
        for (int p = lane; p < 512; p += 32) {
            int n = p & 31, j = p >> 5;
            int gn = tn + n;
            float x01[2] = { 0.f, 0.f };
            if (gn < R3) {
                int dz = gn / R2, dy = (gn / R) % R, dx = gn % R;
#pragma unroll
                for (int u = 0; u < 2; ++u) {
                    int gk = k0 + 2 * j + u;
                    if (gk < Kfull) {
                        int ci = gk / 27, r = gk % 27;
                        int kd = r / 9, kh = (r % 9) / 3, kw = r % 3;
                        int zz = dz + kd - 1, yy = dy + kh - 1, xx = dx + kw - 1;
                        if (zz >= 0 && zz < R && yy >= 0 && yy < R && xx >= 0 && xx < R)
                            x01[u] = Xb[(long long)ci * R3 + (zz * R2 + yy * R + xx)];
                    }
                }
            }
            int ln = (n & 15) + (((j & 7) >= 4) ? 16 : 0);
            int vg = ((j >> 3) << 2) + (j & 3);
            Bpk[n >> 4][ln][vg] = pk2h(x01[0], x01[1]);
        }
        __syncthreads();
        v16h a0 = *(const v16h*)(&Apk[0][lane][0]);
        v16h a1 = *(const v16h*)(&Apk[1][lane][0]);
        v16h b0 = *(const v16h*)(&Bpk[0][lane][0]);
        v16h b1 = *(const v16h*)(&Bpk[1][lane][0]);
        acc00 = __builtin_amdgcn_wmma_f32_16x16x32_f16(false, a0, false, b0, (short)0, acc00, false, false);
        acc01 = __builtin_amdgcn_wmma_f32_16x16x32_f16(false, a0, false, b1, (short)0, acc01, false, false);
        acc10 = __builtin_amdgcn_wmma_f32_16x16x32_f16(false, a1, false, b0, (short)0, acc10, false, false);
        acc11 = __builtin_amdgcn_wmma_f32_16x16x32_f16(false, a1, false, b1, (short)0, acc11, false, false);
        __syncthreads();
    }
    int ncol = lane & 15, hf = lane >> 4;
    v8f accs[2][2] = { { acc00, acc01 }, { acc10, acc11 } };
#pragma unroll
    for (int i = 0; i < 2; ++i)
#pragma unroll
        for (int j2 = 0; j2 < 2; ++j2)
#pragma unroll
            for (int q = 0; q < 8; ++q) {
                int m = tm + i * 16 + q + (hf << 3);
                int n = tn + j2 * 16 + ncol;
                if (m < Cout && n < R3) {
                    float o = accs[i][j2][q];
                    if (bias) o += bias[m];
                    Yb[(long long)m * R3 + n] = o;
                }
            }
}

// --------------------------- utility kernels -------------------------------
__global__ void k_zero(float* p, long long n)
{
    long long i = (long long)blockIdx.x * blockDim.x + threadIdx.x;
    if (i < n) p[i] = 0.f;
}

__global__ void k_transpose(const float* __restrict__ x, float* __restrict__ coords, int N, int B)
{
    int t = blockIdx.x * blockDim.x + threadIdx.x;
    if (t >= B * N) return;
    int b = t / N, n = t % N;
    for (int d = 0; d < 3; ++d)
        coords[((long long)b * 3 + d) * N + n] = x[((long long)b * N + n) * 3 + d];
}

__global__ void k_vox_stats(const float* __restrict__ coords, float* __restrict__ stat, int N)
{
    int b = blockIdx.x;
    const float* cx = coords + (long long)b * 3 * N;
    __shared__ float r0[256], r1[256], r2[256];
    float s0 = 0, s1 = 0, s2 = 0;
    for (int i = threadIdx.x; i < N; i += 256) { s0 += cx[i]; s1 += cx[N + i]; s2 += cx[2 * N + i]; }
    r0[threadIdx.x] = s0; r1[threadIdx.x] = s1; r2[threadIdx.x] = s2;
    __syncthreads();
    for (int o = 128; o > 0; o >>= 1) {
        if (threadIdx.x < o) {
            r0[threadIdx.x] += r0[threadIdx.x + o];
            r1[threadIdx.x] += r1[threadIdx.x + o];
            r2[threadIdx.x] += r2[threadIdx.x + o];
        }
        __syncthreads();
    }
    float mx = r0[0] / N, my = r1[0] / N, mz = r2[0] / N;
    __syncthreads();
    float mm = 0.f;
    for (int i = threadIdx.x; i < N; i += 256) {
        float dx = cx[i] - mx, dy = cx[N + i] - my, dz = cx[2 * N + i] - mz;
        mm = fmaxf(mm, sqrtf(dx * dx + dy * dy + dz * dz));
    }
    r0[threadIdx.x] = mm; __syncthreads();
    for (int o = 128; o > 0; o >>= 1) {
        if (threadIdx.x < o) r0[threadIdx.x] = fmaxf(r0[threadIdx.x], r0[threadIdx.x + o]);
        __syncthreads();
    }
    if (threadIdx.x == 0) {
        stat[b * 4] = mx; stat[b * 4 + 1] = my; stat[b * 4 + 2] = mz;
        stat[b * 4 + 3] = r0[0] * 2.f;
    }
}

__global__ void k_vox_nc(const float* __restrict__ coords, const float* __restrict__ stat,
                         float* __restrict__ nc, int N, int R, int B)
{
    int t = blockIdx.x * blockDim.x + threadIdx.x;
    if (t >= B * N) return;
    int b = t / N, n = t % N;
    float sc = stat[b * 4 + 3];
    for (int d = 0; d < 3; ++d) {
        float v = (coords[((long long)b * 3 + d) * N + n] - stat[b * 4 + d]) / sc + 0.5f;
        v *= (float)R;
        v = fminf(fmaxf(v, 0.f), (float)(R - 1));
        nc[((long long)b * 3 + d) * N + n] = v;
    }
}

__global__ void k_vox_scatter(const float* __restrict__ f, const float* __restrict__ nc,
                              float* __restrict__ vox, float* __restrict__ cnt,
                              int C, int N, int R, int B)
{
    int t = blockIdx.x * blockDim.x + threadIdx.x;
    if (t >= B * N) return;
    int b = t / N, n = t % N;
    int vx = (int)rintf(nc[((long long)b * 3 + 0) * N + n]);
    int vy = (int)rintf(nc[((long long)b * 3 + 1) * N + n]);
    int vz = (int)rintf(nc[((long long)b * 3 + 2) * N + n]);
    vx = vx < 0 ? 0 : (vx > R - 1 ? R - 1 : vx);
    vy = vy < 0 ? 0 : (vy > R - 1 ? R - 1 : vy);
    vz = vz < 0 ? 0 : (vz > R - 1 ? R - 1 : vz);
    int R3 = R * R * R;
    int lin = (vx * R + vy) * R + vz;
    atomicAdd(&cnt[(long long)b * R3 + lin], 1.f);
    for (int c = 0; c < C; ++c)
        atomicAdd(&vox[((long long)b * C + c) * R3 + lin], f[((long long)b * C + c) * N + n]);
}

__global__ void k_vox_div(float* __restrict__ vox, const float* __restrict__ cnt,
                          int C, int R3, int B)
{
    int t = blockIdx.x * blockDim.x + threadIdx.x;
    if (t >= B * R3) return;
    int b = t / R3, lin = t % R3;
    float c = fmaxf(cnt[(long long)b * R3 + lin], 1.f);
    for (int ch = 0; ch < C; ++ch)
        vox[((long long)b * C + ch) * R3 + lin] /= c;
}

// GroupNorm (+optional residual, +optional swish). One block per (batch, group).
__global__ void k_groupnorm(const float* __restrict__ X, const float* __restrict__ RES,
                            float* __restrict__ Y,
                            const float* __restrict__ gamma, const float* __restrict__ beta,
                            int C, int S, int G, int do_swish)
{
    int b = blockIdx.x / G, g = blockIdx.x % G;
    int cpg = C / G;
    long long base = (long long)(b * C + g * cpg) * S;
    int total = cpg * S;
    __shared__ float r0[256], r1[256];
    float s = 0.f, s2 = 0.f;
    for (int i = threadIdx.x; i < total; i += 256) {
        float v = X[base + i];
        if (RES) v += RES[base + i];
        s += v; s2 += v * v;
    }
    r0[threadIdx.x] = s; r1[threadIdx.x] = s2;
    __syncthreads();
    for (int o = 128; o > 0; o >>= 1) {
        if (threadIdx.x < o) { r0[threadIdx.x] += r0[threadIdx.x + o]; r1[threadIdx.x] += r1[threadIdx.x + o]; }
        __syncthreads();
    }
    float mean = r0[0] / total;
    float var  = r1[0] / total - mean * mean;
    float inv  = rsqrtf(var + 1e-5f);
    for (int i = threadIdx.x; i < total; i += 256) {
        float v = X[base + i];
        if (RES) v += RES[base + i];
        int c = g * cpg + i / S;
        float y = (v - mean) * inv * gamma[c] + beta[c];
        if (do_swish) y = swishf(y);
        Y[base + i] = y;
    }
}

__global__ void k_softmax_rows(float* __restrict__ S, int ncols)
{
    float* p = S + (long long)blockIdx.x * ncols;
    __shared__ float red[256];
    float mx = -1e30f;
    for (int i = threadIdx.x; i < ncols; i += 256) mx = fmaxf(mx, p[i]);
    red[threadIdx.x] = mx; __syncthreads();
    for (int o = 128; o > 0; o >>= 1) {
        if (threadIdx.x < o) red[threadIdx.x] = fmaxf(red[threadIdx.x], red[threadIdx.x + o]);
        __syncthreads();
    }
    mx = red[0]; __syncthreads();
    float s = 0.f;
    for (int i = threadIdx.x; i < ncols; i += 256) { float e = expf(p[i] - mx); p[i] = e; s += e; }
    red[threadIdx.x] = s; __syncthreads();
    for (int o = 128; o > 0; o >>= 1) {
        if (threadIdx.x < o) red[threadIdx.x] += red[threadIdx.x + o];
        __syncthreads();
    }
    float inv = 1.f / red[0];
    for (int i = threadIdx.x; i < ncols; i += 256) p[i] *= inv;
}

__global__ void k_se_mean(const float* __restrict__ X, float* __restrict__ s, int S)
{
    int bc = blockIdx.x;
    const float* p = X + (long long)bc * S;
    __shared__ float red[256];
    float a = 0.f;
    for (int i = threadIdx.x; i < S; i += 256) a += p[i];
    red[threadIdx.x] = a; __syncthreads();
    for (int o = 128; o > 0; o >>= 1) {
        if (threadIdx.x < o) red[threadIdx.x] += red[threadIdx.x + o];
        __syncthreads();
    }
    if (threadIdx.x == 0) s[bc] = red[0] / S;
}

__global__ void k_se_fc(const float* __restrict__ s, const float* __restrict__ w1,
                        const float* __restrict__ w2, float* __restrict__ s2, int C, int Cr)
{
    int b = blockIdx.x, t = threadIdx.x;
    __shared__ float sh[64], sh1[16];
    if (t < C) sh[t] = s[b * C + t];
    __syncthreads();
    if (t < Cr) {
        float a = 0.f;
        for (int c = 0; c < C; ++c) a += w1[t * C + c] * sh[c];
        sh1[t] = swishf(a);
    }
    __syncthreads();
    if (t < C) {
        float a = 0.f;
        for (int r = 0; r < Cr; ++r) a += w2[t * Cr + r] * sh1[r];
        s2[b * C + t] = 1.f / (1.f + expf(-a));
    }
}

__global__ void k_se_scale(float* __restrict__ X, const float* __restrict__ s,
                           long long total, int S)
{
    long long i = (long long)blockIdx.x * blockDim.x + threadIdx.x;
    if (i < total) X[i] *= s[i / S];
}

__global__ void k_devox_add(const float* __restrict__ gr, const float* __restrict__ nc,
                            const float* __restrict__ pt, float* __restrict__ out,
                            int C, int R, int N, int B)
{
    int t = blockIdx.x * blockDim.x + threadIdx.x;
    if (t >= B * N) return;
    int b = t / N, n = t % N;
    float fx = nc[((long long)b * 3 + 0) * N + n];
    float fy = nc[((long long)b * 3 + 1) * N + n];
    float fz = nc[((long long)b * 3 + 2) * N + n];
    int x0 = (int)floorf(fx), y0 = (int)floorf(fy), z0 = (int)floorf(fz);
    int x1 = x0 + 1 < R ? x0 + 1 : R - 1;
    int y1 = y0 + 1 < R ? y0 + 1 : R - 1;
    int z1 = z0 + 1 < R ? z0 + 1 : R - 1;
    float tx = fx - (float)x0, ty = fy - (float)y0, tz = fz - (float)z0;
    int R3 = R * R * R;
    int li[8]; float wt[8];
    li[0] = (x0 * R + y0) * R + z0; wt[0] = (1 - tx) * (1 - ty) * (1 - tz);
    li[1] = (x0 * R + y0) * R + z1; wt[1] = (1 - tx) * (1 - ty) * tz;
    li[2] = (x0 * R + y1) * R + z0; wt[2] = (1 - tx) * ty * (1 - tz);
    li[3] = (x0 * R + y1) * R + z1; wt[3] = (1 - tx) * ty * tz;
    li[4] = (x1 * R + y0) * R + z0; wt[4] = tx * (1 - ty) * (1 - tz);
    li[5] = (x1 * R + y0) * R + z1; wt[5] = tx * (1 - ty) * tz;
    li[6] = (x1 * R + y1) * R + z0; wt[6] = tx * ty * (1 - tz);
    li[7] = (x1 * R + y1) * R + z1; wt[7] = tx * ty * tz;
    for (int c = 0; c < C; ++c) {
        const float* g = gr + ((long long)b * C + c) * R3;
        float a = 0.f;
#pragma unroll
        for (int j = 0; j < 8; ++j) a += g[li[j]] * wt[j];
        out[((long long)b * C + c) * N + n] = a + pt[((long long)b * C + c) * N + n];
    }
}

// Farthest point sampling: one block per batch, dist kept in LDS.
__global__ void k_fps(const float* __restrict__ coords, float* __restrict__ centers, int N, int M)
{
    int b = blockIdx.x;
    const float* cx = coords + (long long)b * 3 * N;
    float* ctr = centers + (long long)b * 3 * M;
    __shared__ float dist[2048];
    __shared__ float bv[256];
    __shared__ int   bi[256];
    __shared__ float last[3];
    for (int i = threadIdx.x; i < N; i += 256) dist[i] = 1e10f;
    if (threadIdx.x == 0) {
        last[0] = cx[0]; last[1] = cx[N]; last[2] = cx[2 * N];
        ctr[0] = last[0]; ctr[M] = last[1]; ctr[2 * M] = last[2];
    }
    __syncthreads();
    for (int it = 1; it < M; ++it) {
        float lx = last[0], ly = last[1], lz = last[2];
        float best = -1.f; int besti = 0;
        for (int i = threadIdx.x; i < N; i += 256) {
            float dx = cx[i] - lx, dy = cx[N + i] - ly, dz = cx[2 * N + i] - lz;
            float d = dx * dx + dy * dy + dz * dz;
            float dm = fminf(dist[i], d);
            dist[i] = dm;
            if (dm > best) { best = dm; besti = i; }
        }
        bv[threadIdx.x] = best; bi[threadIdx.x] = besti;
        __syncthreads();
        for (int o = 128; o > 0; o >>= 1) {
            if (threadIdx.x < o) {
                if (bv[threadIdx.x + o] > bv[threadIdx.x] ||
                    (bv[threadIdx.x + o] == bv[threadIdx.x] && bi[threadIdx.x + o] < bi[threadIdx.x])) {
                    bv[threadIdx.x] = bv[threadIdx.x + o];
                    bi[threadIdx.x] = bi[threadIdx.x + o];
                }
            }
            __syncthreads();
        }
        if (threadIdx.x == 0) {
            int nx = bi[0];
            last[0] = cx[nx]; last[1] = cx[N + nx]; last[2] = cx[2 * N + nx];
            ctr[it] = last[0]; ctr[M + it] = last[1]; ctr[2 * M + it] = last[2];
        }
        __syncthreads();
    }
}

__global__ void k_ball_query(const float* __restrict__ centers, const float* __restrict__ coords,
                             int* __restrict__ idx, int N, int M, int K, float r2, int B)
{
    int t = blockIdx.x * blockDim.x + threadIdx.x;
    if (t >= B * M) return;
    int b = t / M, m = t % M;
    const float* cx = coords + (long long)b * 3 * N;
    float px = centers[((long long)b * 3 + 0) * M + m];
    float py = centers[((long long)b * 3 + 1) * M + m];
    float pz = centers[((long long)b * 3 + 2) * M + m];
    int slots[32]; int cnt = 0;
    for (int n = 0; n < N; ++n) {
        float dx = cx[n] - px, dy = cx[N + n] - py, dz = cx[2 * N + n] - pz;
        if (dx * dx + dy * dy + dz * dz < r2) {
            if (cnt < K) slots[cnt] = n;
            cnt++;
        }
    }
    int first = (cnt > 0) ? slots[0] : 0;
    int* op = idx + ((long long)b * M + m) * K;
    for (int k = 0; k < K; ++k) op[k] = (k < cnt && k < K) ? slots[k] : first;
}

__global__ void k_group_concat(const float* __restrict__ f, const float* __restrict__ coords,
                               const float* __restrict__ centers, const int* __restrict__ idx,
                               float* __restrict__ nb, int C, int N, int M, int K, int B)
{
    int t = blockIdx.x * blockDim.x + threadIdx.x;
    if (t >= B * M * K) return;
    int k = t % K, m = (t / K) % M, b = t / (M * K);
    int g = idx[((long long)b * M + m) * K + k];
    long long MK = (long long)M * K;
    long long nbbase = (long long)b * (C + 3) * MK;
    long long pos = (long long)m * K + k;
    for (int d = 0; d < 3; ++d)
        nb[nbbase + (long long)d * MK + pos] =
            coords[((long long)b * 3 + d) * N + g] - centers[((long long)b * 3 + d) * M + m];
    for (int c = 0; c < C; ++c)
        nb[nbbase + (long long)(3 + c) * MK + pos] = f[((long long)b * C + c) * N + g];
}

__global__ void k_max_lastdim(const float* __restrict__ X, float* __restrict__ Y, int rows, int Kd)
{
    int r = blockIdx.x * blockDim.x + threadIdx.x;
    if (r >= rows) return;
    const float* p = X + (long long)r * Kd;
    float m = p[0];
    for (int k = 1; k < Kd; ++k) m = fmaxf(m, p[k]);
    Y[r] = m;
}

__global__ void k_split_out(const float* __restrict__ tmp, float* __restrict__ out)
{
    int t = blockIdx.x * blockDim.x + threadIdx.x;
    if (t >= 2048) return;
    int p = t / 1024, r = t % 1024, b = r / 128, j = r % 128;
    out[t] = tmp[(long long)(p * 128 + j) * 8 + b];  // tmp is [256 outputs, 8 batches]
}

// --------------------------- host orchestration ----------------------------
struct PvP {
    const float *w1, *b1, *g1, *be1, *w2, *b2, *g2, *be2, *sew1, *sew2,
                *ptw, *ptb, *ptg, *ptbe,
                *wq, *bq, *wk, *bk, *wv, *bv, *wo, *bo, *ag, *abe;
};

struct Scratch {
    float *nc, *stat, *vox, *cnt, *h1, *h2, *ptbuf, *ses, *ses2;
    float *qb, *kb, *vb, *hhb, *hh2b, *scores;
};

static void gemm(hipStream_t st, const float* A, int lda, int tA, long long sA,
                 const float* Bm, int ldb, int tB, long long sB,
                 float* C, int ldc, long long sC, const float* bias,
                 int M, int N, int K, int batches)
{
    dim3 g(idiv(N, 32), idiv(M, 32), batches);
    if (tA == 0 && tB == 0)
        k_wmma_gemm_t<0, 0><<<g, 32, 0, st>>>(A, Bm, C, bias, M, N, K, lda, ldb, ldc, sA, sB, sC);
    else if (tA == 1 && tB == 0)
        k_wmma_gemm_t<1, 0><<<g, 32, 0, st>>>(A, Bm, C, bias, M, N, K, lda, ldb, ldc, sA, sB, sC);
    else
        k_wmma_gemm_t<0, 1><<<g, 32, 0, st>>>(A, Bm, C, bias, M, N, K, lda, ldb, ldc, sA, sB, sC);
}

static void gnorm(hipStream_t st, const float* X, const float* RES, float* Y,
                  const float* g, const float* be, int Bn, int C, int S, int sw)
{
    k_groupnorm<<<dim3(Bn * 8), 256, 0, st>>>(X, RES, Y, g, be, C, S, 8, sw);
}

static void pvconv_run(hipStream_t st, const Scratch& w, const PvP& p,
                       const float* f, const float* coords, float* outf,
                       int Bn, int Cin, int Cout, int N, int R, bool attn)
{
    int R3 = R * R * R;
    // voxelize
    k_vox_stats<<<Bn, 256, 0, st>>>(coords, w.stat, N);
    k_vox_nc<<<idiv(Bn * N, 256), 256, 0, st>>>(coords, w.stat, w.nc, N, R, Bn);
    k_zero<<<idiv(Bn * Cin * R3, 256), 256, 0, st>>>(w.vox, (long long)Bn * Cin * R3);
    k_zero<<<idiv(Bn * R3, 256), 256, 0, st>>>(w.cnt, (long long)Bn * R3);
    k_vox_scatter<<<idiv(Bn * N, 256), 256, 0, st>>>(f, w.nc, w.vox, w.cnt, Cin, N, R, Bn);
    k_vox_div<<<idiv(Bn * R3, 256), 256, 0, st>>>(w.vox, w.cnt, Cin, R3, Bn);
    // conv1 + GN + swish
    {
        dim3 g(idiv(R3, 32), idiv(Cout, 32), Bn);
        k_conv3d_wmma<<<g, 32, 0, st>>>(p.w1, p.b1, w.vox, w.h1, Cout, Cin, R);
    }
    gnorm(st, w.h1, nullptr, w.h1, p.g1, p.be1, Bn, Cout, R3, 1);
    // conv2 + GN (swish only when no attention)
    {
        dim3 g(idiv(R3, 32), idiv(Cout, 32), Bn);
        k_conv3d_wmma<<<g, 32, 0, st>>>(p.w2, p.b2, w.h1, w.h2, Cout, Cout, R);
    }
    gnorm(st, w.h2, nullptr, w.h2, p.g2, p.be2, Bn, Cout, R3, attn ? 0 : 1);
    if (attn) {
        long long sBh = (long long)Cout * R3;
        gemm(st, p.wq, Cout, 0, 0, w.h2, R3, 0, sBh, w.qb, R3, sBh, p.bq, Cout, R3, Cout, Bn);
        gemm(st, p.wk, Cout, 0, 0, w.h2, R3, 0, sBh, w.kb, R3, sBh, p.bk, Cout, R3, Cout, Bn);
        gemm(st, p.wv, Cout, 0, 0, w.h2, R3, 0, sBh, w.vb, R3, sBh, p.bv, Cout, R3, Cout, Bn);
        for (int b = 0; b < Bn; ++b) {
            const float* qb = w.qb + (long long)b * sBh;
            const float* kb = w.kb + (long long)b * sBh;
            const float* vb = w.vb + (long long)b * sBh;
            float*       hb = w.hhb + (long long)b * sBh;
            // scores[m,n] = sum_c q[c,m] k[c,n]
            gemm(st, qb, R3, 1, 0, kb, R3, 0, 0, w.scores, R3, 0, nullptr, R3, R3, Cout, 1);
            k_softmax_rows<<<R3, 256, 0, st>>>(w.scores, R3);
            // hh[c,m] = sum_n v[c,n] w[m,n]
            gemm(st, vb, R3, 0, 0, w.scores, R3, 1, 0, hb, R3, 0, nullptr, Cout, R3, R3, 1);
        }
        gemm(st, p.wo, Cout, 0, 0, w.hhb, R3, 0, sBh, w.hh2b, R3, sBh, p.bo, Cout, R3, Cout, Bn);
        gnorm(st, w.hh2b, w.h2, w.h2, p.ag, p.abe, Bn, Cout, R3, 1);   // residual + GN + swish
    }
    // SE
    k_se_mean<<<Bn * Cout, 256, 0, st>>>(w.h2, w.ses, R3);
    k_se_fc<<<Bn, 64, 0, st>>>(w.ses, p.sew1, p.sew2, w.ses2, Cout, Cout / 8);
    {
        long long tot = (long long)Bn * Cout * R3;
        k_se_scale<<<dim3((unsigned)((tot + 255) / 256)), 256, 0, st>>>(w.h2, w.ses2, tot, R3);
    }
    // point branch MLP + GN + swish
    gemm(st, p.ptw, Cin, 0, 0, f, N, 0, (long long)Cin * N, w.ptbuf, N, (long long)Cout * N,
         p.ptb, Cout, N, Cin, Bn);
    gnorm(st, w.ptbuf, nullptr, w.ptbuf, p.ptg, p.ptbe, Bn, Cout, N, 1);
    // devoxelize + add
    k_devox_add<<<idiv(Bn * N, 256), 256, 0, st>>>(w.h2, w.nc, w.ptbuf, outf, Cout, R, N, Bn);
}

static void sa_run(hipStream_t st, float* nbA, float* nbB, int* idx,
                   const float* f, const float* coords, float* centers, float* outf,
                   const float* w0, const float* b0, const float* g0, const float* be0, int oc0,
                   const float* w1, const float* b1, const float* g1, const float* be1, int oc1,
                   int Bn, int Cin, int N, int M, int K, float radius)
{
    k_fps<<<Bn, 256, 0, st>>>(coords, centers, N, M);
    k_ball_query<<<idiv(Bn * M, 256), 256, 0, st>>>(centers, coords, idx, N, M, K,
                                                    radius * radius, Bn);
    k_group_concat<<<idiv(Bn * M * K, 256), 256, 0, st>>>(f, coords, centers, idx, nbA,
                                                          Cin, N, M, K, Bn);
    int MK = M * K, Cg = Cin + 3;
    gemm(st, w0, Cg, 0, 0, nbA, MK, 0, (long long)Cg * MK, nbB, MK, (long long)oc0 * MK,
         b0, oc0, MK, Cg, Bn);
    gnorm(st, nbB, nullptr, nbB, g0, be0, Bn, oc0, MK, 1);
    gemm(st, w1, oc0, 0, 0, nbB, MK, 0, (long long)oc0 * MK, nbA, MK, (long long)oc1 * MK,
         b1, oc1, MK, oc0, Bn);
    gnorm(st, nbA, nullptr, nbA, g1, be1, Bn, oc1, MK, 1);
    k_max_lastdim<<<idiv(Bn * oc1 * M, 256), 256, 0, st>>>(nbA, outf, Bn * oc1 * M, K);
}

extern "C" void kernel_launch(void* const* d_in, const int* in_sizes, int n_in,
                              void* d_out, int out_size, void* d_ws, size_t ws_size,
                              hipStream_t stream)
{
    (void)in_sizes; (void)n_in; (void)out_size; (void)ws_size;
    const int Bn = 8, N0 = 2048;
    auto F = [&](int i) { return (const float*)d_in[i]; };

    // Bump allocator over the workspace (~268 MB total).
    char* base = (char*)d_ws;
    size_t off = 0;
    auto alloc_f = [&](long long n) -> float* {
        float* p = (float*)(base + off);
        off += ((size_t)n * 4 + 255) & ~(size_t)255;
        return p;
    };
    auto alloc_i = [&](long long n) -> int* {
        int* p = (int*)(base + off);
        off += ((size_t)n * 4 + 255) & ~(size_t)255;
        return p;
    };

    float* coords = alloc_f((long long)Bn * 3 * N0);
    float* feat_a = alloc_f((long long)Bn * 32 * N0);
    float* feat_b = alloc_f((long long)Bn * 32 * N0);
    Scratch w;
    w.nc     = alloc_f((long long)Bn * 3 * N0);
    w.stat   = alloc_f(Bn * 4);
    w.vox    = alloc_f((long long)Bn * 32 * 32768);
    w.cnt    = alloc_f((long long)Bn * 32768);
    w.h1     = alloc_f((long long)Bn * 32 * 32768);
    w.h2     = alloc_f((long long)Bn * 32 * 32768);
    w.ptbuf  = alloc_f((long long)Bn * 32 * N0);
    w.ses    = alloc_f(Bn * 32);
    w.ses2   = alloc_f(Bn * 32);
    w.qb     = alloc_f((long long)Bn * 32 * 4096);
    w.kb     = alloc_f((long long)Bn * 32 * 4096);
    w.vb     = alloc_f((long long)Bn * 32 * 4096);
    w.hhb    = alloc_f((long long)Bn * 32 * 4096);
    w.hh2b   = alloc_f((long long)Bn * 32 * 4096);
    w.scores = alloc_f((long long)4096 * 4096);     // per-batch, reused sequentially
    float* centersA = alloc_f((long long)Bn * 3 * 1024);
    float* centersB = alloc_f((long long)Bn * 3 * 1024);
    int*   idx      = alloc_i((long long)Bn * 1024 * 32);
    float* nbA      = alloc_f((long long)Bn * 35 * 1024 * 32);
    float* nbB      = alloc_f((long long)Bn * 32 * 1024 * 32);
    float* fmaxb    = alloc_f(Bn * 64);
    float* ftmp     = alloc_f(256 * 8);

    // coords = transpose(x)  (B,2048,3) -> (B,3,2048); initial features = coords
    k_transpose<<<idiv(Bn * N0, 256), 256, 0, stream>>>(F(0), coords, N0, Bn);

    // Param flattening assumed to follow setup_inputs() insertion order.
    PvP p0 = { F(1), F(2), F(3), F(4), F(5), F(6), F(7), F(8), F(9), F(10),
               F(11), F(12), F(13), F(14) };
    pvconv_run(stream, w, p0, coords, coords, feat_a, Bn, 3, 32, N0, 32, false);

    PvP p1 = { F(15), F(16), F(17), F(18), F(19), F(20), F(21), F(22), F(23), F(24),
               F(25), F(26), F(27), F(28) };
    pvconv_run(stream, w, p1, feat_a, coords, feat_b, Bn, 32, 32, N0, 32, false);

    // sa0: N=2048 -> M=1024, r=0.1, K=32, chans [32,32]
    sa_run(stream, nbA, nbB, idx, feat_b, coords, centersA, feat_a,
           F(29), F(30), F(31), F(32), 32, F(33), F(34), F(35), F(36), 32,
           Bn, 32, N0, 1024, 32, 0.1f);

    // pv2 with attention, R=16
    PvP p2 = { F(37), F(38), F(39), F(40), F(41), F(42), F(43), F(44), F(45), F(46),
               F(47), F(48), F(49), F(50),
               F(51), F(52), F(53), F(54), F(55), F(56), F(57), F(58), F(59), F(60) };
    pvconv_run(stream, w, p2, feat_a, centersA, feat_b, Bn, 32, 32, 1024, 16, true);

    // sa1: N=1024 -> M=256, r=0.2, K=32, chans [32,64]
    sa_run(stream, nbA, nbB, idx, feat_b, centersA, centersB, feat_a,
           F(61), F(62), F(63), F(64), 32, F(65), F(66), F(67), F(68), 64,
           Bn, 32, 1024, 256, 32, 0.2f);

    // global max over points, final MLP, split halves
    k_max_lastdim<<<idiv(Bn * 64, 256), 256, 0, stream>>>(feat_a, fmaxb, Bn * 64, 256);
    // ftmp[o,b] = sum_c mlp_w[o,c] * fmax[b,c] + mlp_b[o]
    gemm(stream, F(69), 64, 0, 0, fmaxb, 64, 1, 0, ftmp, 8, 0, F(70), 256, 8, 64, 1);
    k_split_out<<<idiv(2048, 256), 256, 0, stream>>>(ftmp, (float*)d_out);
}
